// CRvNN_balanced_32341103738883
// MI455X (gfx1250) — matrix operand order repaired and checked by reading, not compiled, not run on hardware
//
#include <hip/hip_runtime.h>
#include <hip/hip_bf16.h>

// ---------------------------------------------------------------------------
// Types for gfx1250 WMMA (wave32): V_WMMA_F32_16X16X32_BF16
// ---------------------------------------------------------------------------
typedef __attribute__((ext_vector_type(16))) __bf16 bf16x16;
typedef __attribute__((ext_vector_type(8)))  __bf16 bf16x8;
typedef __attribute__((ext_vector_type(8)))  float  f32x8;

#if defined(__HIP_DEVICE_COMPILE__)
typedef __attribute__((address_space(3))) void lds_void_t;
#endif

#define LDSS 40   // LDS row stride in bf16 elements (32 data + 8 pad, keeps 16B align)

static __device__ inline bf16x16 combine8(bf16x8 lo, bf16x8 hi) {
  bf16x16 r;
#pragma unroll
  for (int i = 0; i < 8; ++i) { r[i] = lo[i]; r[i + 8] = hi[i]; }
  return r;
}

// Async direct global->LDS 16B copy (gfx1250 GLOBAL_LOAD_ASYNC_TO_LDS_B128,
// tracked with ASYNCcnt). vdst = per-lane 32-bit LDS byte address.
static __device__ inline void async_copy_b128(const void* gsrc, void* ldst) {
#if defined(__HIP_DEVICE_COMPILE__)
  unsigned lds_addr = (unsigned)(uintptr_t)(lds_void_t*)ldst;
  unsigned long long gaddr = (unsigned long long)(uintptr_t)gsrc;
  asm volatile("global_load_async_to_lds_b128 %0, %1, off"
               :: "v"(lds_addr), "v"(gaddr)
               : "memory");
#else
  (void)gsrc; (void)ldst;
#endif
}

static __device__ inline void wait_async_le4() {
#if defined(__HIP_DEVICE_COMPILE__)
  asm volatile("s_wait_asynccnt 0x4" ::: "memory");
#endif
}
static __device__ inline void wait_async_0() {
#if defined(__HIP_DEVICE_COMPILE__)
  asm volatile("s_wait_asynccnt 0x0" ::: "memory");
#endif
}

// ---------------------------------------------------------------------------
// GEMM: C[M x Nout] = A[M x K] (bf16, row-major) @ Bt[Nout x K] (bf16, K-major)
// MODE 0: out = bf16( relu(acc + bias) )     (hidden layer h)
// MODE 1: out = f32 ( acc + bias )           (contents / init projection)
// Block: 256 threads = 8 waves; 128x128 output tile; K stepped by 32.
// Wave grid 2(M) x 4(N); each wave computes 4x2 = 8 WMMA 16x16 tiles.
// LDS double-buffered; tiles staged with async-to-LDS (ASYNCcnt pipeline):
//   issue tile k+1 -> s_wait_asynccnt<=4 (tile k resident) -> barrier -> WMMA.
// Each wave issues exactly 4 async b128 ops per tile; async loads complete
// in order, so waiting ASYNCcnt<=4 retires the previous tile's 4 ops.
// ---------------------------------------------------------------------------
template <int MODE>
__global__ __launch_bounds__(256, 2)
void gemm_bf16_wmma(const __bf16* __restrict__ A, const __bf16* __restrict__ Bt,
                    const float* __restrict__ bias, void* __restrict__ outp,
                    int M, int K, int Nout) {
  __shared__ __align__(16) __bf16 lA[2][128 * LDSS];
  __shared__ __align__(16) __bf16 lB[2][128 * LDSS];

  const int tid   = threadIdx.x;
  const int n0    = blockIdx.x * 128;
  const int m0    = blockIdx.y * 128;
  const int wave  = tid >> 5;
  const int lane  = tid & 31;
  const int wm    = wave & 1;   // 0..1  -> 64-row strip
  const int wn    = wave >> 1;  // 0..3  -> 32-col strip
  const int lhalf = lane >> 4;  // 0 or 1
  const int l16   = lane & 15;

  // Per-thread staging coordinates: 512 16B chunks per 128x32 tile, 2/thread.
  const int ch0  = tid * 2;
  const int row0 = ch0 >> 2, seg0 = ch0 & 3;
  const int row1 = (ch0 + 1) >> 2, seg1 = (ch0 + 1) & 3;
  int gr0 = m0 + row0; if (gr0 >= M) gr0 = M - 1;   // clamp OOB rows (store-guarded)
  int gr1 = m0 + row1; if (gr1 >= M) gr1 = M - 1;

  f32x8 acc[4][2];
#pragma unroll
  for (int t = 0; t < 4; ++t)
#pragma unroll
    for (int u = 0; u < 2; ++u)
#pragma unroll
      for (int e = 0; e < 8; ++e) acc[t][u][e] = 0.0f;

  // ---- issue async staging of one 128x32 A tile + B tile (4 b128 ops/thread)
  auto stage = [&](int k0, int buf) {
    async_copy_b128(A  + (size_t)gr0 * K + k0 + seg0 * 8,
                    &lA[buf][row0 * LDSS + seg0 * 8]);
    async_copy_b128(Bt + (size_t)(n0 + row0) * K + k0 + seg0 * 8,
                    &lB[buf][row0 * LDSS + seg0 * 8]);
    async_copy_b128(A  + (size_t)gr1 * K + k0 + seg1 * 8,
                    &lA[buf][row1 * LDSS + seg1 * 8]);
    async_copy_b128(Bt + (size_t)(n0 + row1) * K + k0 + seg1 * 8,
                    &lB[buf][row1 * LDSS + seg1 * 8]);
  };

  stage(0, 0);
  int buf = 0;

  for (int k0 = 0; k0 < K; k0 += 32) {
    const bool more = (k0 + 32) < K;
    if (more) {
      stage(k0 + 32, buf ^ 1);   // prefetch next tile into other buffer
      wait_async_le4();          // previous 4 ops (this tile) have retired
    } else {
      wait_async_0();
    }
    __syncthreads();             // whole tile resident across all 8 waves

    // ---- gather fragments per CDNA5 16-bit WMMA VGPR layouts
    // A 16x32: lanes<16 -> K {0..7,16..23}; lanes>=16 -> K {8..15,24..31}
    const int akb = lhalf ? 8 : 0;
    bf16x16 af[4];
#pragma unroll
    for (int t = 0; t < 4; ++t) {
      int mr = wm * 64 + t * 16 + l16;
      bf16x8 lo = *(const bf16x8*)&lA[buf][mr * LDSS + akb];
      bf16x8 hi = *(const bf16x8*)&lA[buf][mr * LDSS + akb + 16];
      af[t] = combine8(lo, hi);
    }
    // B 32x16: lanes<16 -> K {0..7,16..23}; lanes>=16 -> K {8..15,24..31} (Bt tile K-contiguous)
    const int bkb = lhalf * 8;
    bf16x16 bfr[2];
#pragma unroll
    for (int u = 0; u < 2; ++u) {
      int nr = wn * 32 + u * 16 + l16;
      bf16x8 lo = *(const bf16x8*)&lB[buf][nr * LDSS + bkb];
      bf16x8 hi = *(const bf16x8*)&lB[buf][nr * LDSS + bkb + 16];
      bfr[u] = combine8(lo, hi);
    }

#pragma unroll
    for (int t = 0; t < 4; ++t)
#pragma unroll
      for (int u = 0; u < 2; ++u)
        acc[t][u] = __builtin_amdgcn_wmma_f32_16x16x32_bf16(
            false, af[t], false, bfr[u], (short)0, acc[t][u], false, false);

    __syncthreads();             // all reads done before next async overwrite
    buf ^= 1;
  }

  // ---- epilogue: C/D layout: VGPR v -> M = v + 8*lhalf, N = l16
#pragma unroll
  for (int t = 0; t < 4; ++t) {
#pragma unroll
    for (int u = 0; u < 2; ++u) {
      int col  = n0 + wn * 32 + u * 16 + l16;
      float bv = bias[col];
#pragma unroll
      for (int v = 0; v < 8; ++v) {
        int row = m0 + wm * 64 + t * 16 + lhalf * 8 + v;
        if (row < M) {
          float val = acc[t][u][v] + bv;
          if (MODE == 0) {
            ((__bf16*)outp)[(size_t)row * Nout + col] = (__bf16)fmaxf(val, 0.0f);
          } else {
            ((float*)outp)[(size_t)row * Nout + col] = val;
          }
        }
      }
    }
  }
}

// ---------------------------------------------------------------------------
// Tiled transpose + fp32 -> bf16 convert:  in[R][C] -> out[C][R]
// ---------------------------------------------------------------------------
__global__ __launch_bounds__(256)
void transpose_convert(const float* __restrict__ in, __bf16* __restrict__ out,
                       int R, int C) {
  __shared__ float tile[32][33];
  int cb = blockIdx.x * 32, rb = blockIdx.y * 32;
  int tx = threadIdx.x, ty = threadIdx.y;   // block (32, 8)
#pragma unroll
  for (int i = 0; i < 32; i += 8)
    tile[ty + i][tx] = in[(size_t)(rb + ty + i) * C + cb + tx];
  __syncthreads();
#pragma unroll
  for (int i = 0; i < 32; i += 8)
    out[(size_t)(cb + ty + i) * R + rb + tx] = (__bf16)tile[tx][ty + i];
}

// ---------------------------------------------------------------------------
// x0 = bf16(sequence * mask_row)
// ---------------------------------------------------------------------------
__global__ __launch_bounds__(256)
void prep_x0(const float* __restrict__ seq, const float* __restrict__ mask,
             __bf16* __restrict__ A0, int Dd) {
  int row = blockIdx.x;
  float m = mask[row];
  size_t base = (size_t)row * Dd;
  for (int d = threadIdx.x; d < Dd; d += blockDim.x)
    A0[base + d] = (__bf16)(seq[base + d] * m);
}

__global__ __launch_bounds__(256)
void copy_mask(const float* __restrict__ src, float* __restrict__ dst, int n) {
  int i = blockIdx.x * blockDim.x + threadIdx.x;
  if (i < n) dst[i] = src[i];
}

// ---------------------------------------------------------------------------
// block-wide sum reduction of (a, b) across 256 threads (wave32 shuffles + LDS)
// ---------------------------------------------------------------------------
static __device__ inline void block_reduce2(float& a, float& b) {
  __shared__ float sa[8], sb[8];
#pragma unroll
  for (int off = 16; off > 0; off >>= 1) {
    a += __shfl_xor(a, off);
    b += __shfl_xor(b, off);
  }
  int wid = threadIdx.x >> 5, lane = threadIdx.x & 31;
  if (lane == 0) { sa[wid] = a; sb[wid] = b; }
  __syncthreads();
  if (wid == 0) {
    a = (lane < 8) ? sa[lane] : 0.0f;
    b = (lane < 8) ? sb[lane] : 0.0f;
#pragma unroll
    for (int off = 4; off > 0; off >>= 1) {
      a += __shfl_xor(a, off);
      b += __shfl_xor(b, off);
    }
    if (lane == 0) { sa[0] = a; sb[0] = b; }
  }
  __syncthreads();
  a = sa[0]; b = sb[0];
}

// ---------------------------------------------------------------------------
// LN after init GEMM:  x = bf16( LN(tmp_row) * mask_row )
// ---------------------------------------------------------------------------
__global__ __launch_bounds__(256)
void ln_init(const float* __restrict__ tmp, const float* __restrict__ mask,
             const float* __restrict__ g, const float* __restrict__ b,
             __bf16* __restrict__ xout, int Dd) {
  int row = blockIdx.x;
  const float* x = tmp + (size_t)row * Dd;
  float vals[4], s = 0.0f, s2 = 0.0f;
#pragma unroll
  for (int q = 0; q < 4; ++q) {
    int d = threadIdx.x + q * 256;
    float v = x[d];
    vals[q] = v; s += v; s2 += v * v;
  }
  block_reduce2(s, s2);
  float mu  = s / Dd;
  float var = s2 / Dd - mu * mu;
  float inv = rsqrtf(var + 1e-5f);
  float mk  = mask[row];
#pragma unroll
  for (int q = 0; q < 4; ++q) {
    int d = threadIdx.x + q * 256;
    float y = (vals[q] - mu) * inv * g[d] + b[d];
    xout[(size_t)row * Dd + d] = (__bf16)(y * mk);
  }
}

static __device__ inline float sigmoidf_(float x) { return 1.0f / (1.0f + __expf(-x)); }

// ---------------------------------------------------------------------------
// compose: gates + LN + mask blend. One block per output row j.
//   contents row j: [f1 | f2 | i | parent], each Dd wide
//   l = xcur[2j], r = xcur[2j+1];  z = sig(f1)*l + sig(f2)*r + sig(i)*p
//   y = LN(z)*g + b;  out = mr*y + (1-mr)*l; mnext[j] = ml
// If out_f32 != nullptr also writes fp32 at [j*Dd] and [out2 + j*Dd].
// ---------------------------------------------------------------------------
__global__ __launch_bounds__(256)
void compose(const float* __restrict__ contents, const __bf16* __restrict__ xcur,
             const float* __restrict__ mcur, const float* __restrict__ g,
             const float* __restrict__ b, __bf16* __restrict__ xnext,
             float* __restrict__ mnext, float* __restrict__ out_f32,
             int Dd, int out2) {
  int j = blockIdx.x;
  const float*  c  = contents + (size_t)j * 4 * Dd;
  const __bf16* lr = xcur + (size_t)(2 * j) * Dd;
  const __bf16* rr = xcur + (size_t)(2 * j + 1) * Dd;
  float ml = mcur[2 * j], mr = mcur[2 * j + 1];

  float lz[4], zz[4], s = 0.0f, s2 = 0.0f;
#pragma unroll
  for (int q = 0; q < 4; ++q) {
    int d = threadIdx.x + q * 256;
    float l  = (float)lr[d];
    float r  = (float)rr[d];
    float f1 = sigmoidf_(c[d]);
    float f2 = sigmoidf_(c[Dd + d]);
    float ii = sigmoidf_(c[2 * Dd + d]);
    float p  = c[3 * Dd + d];
    float z  = f1 * l + f2 * r + ii * p;
    lz[q] = l; zz[q] = z; s += z; s2 += z * z;
  }
  block_reduce2(s, s2);
  float mu  = s / Dd;
  float var = s2 / Dd - mu * mu;
  float inv = rsqrtf(var + 1e-5f);
#pragma unroll
  for (int q = 0; q < 4; ++q) {
    int d = threadIdx.x + q * 256;
    float y = (zz[q] - mu) * inv * g[d] + b[d];
    float o = mr * y + (1.0f - mr) * lz[q];
    xnext[(size_t)j * Dd + d] = (__bf16)o;
    if (out_f32) {
      out_f32[(size_t)j * Dd + d]                = o;
      out_f32[(size_t)out2 + (size_t)j * Dd + d] = o;
    }
  }
  if (threadIdx.x == 0) mnext[j] = ml;
}

// ---------------------------------------------------------------------------
// Host-side orchestration
// ---------------------------------------------------------------------------
extern "C" void kernel_launch(void* const* d_in, const int* in_sizes, int n_in,
                              void* d_out, int out_size, void* d_ws, size_t ws_size,
                              hipStream_t stream) {
  (void)in_sizes; (void)n_in; (void)out_size; (void)ws_size;
  const float* seq   = (const float*)d_in[0];
  const float* imask = (const float*)d_in[1];
  const float* Wi    = (const float*)d_in[2];
  const float* bi    = (const float*)d_in[3];
  const float* W1    = (const float*)d_in[4];
  const float* b1    = (const float*)d_in[5];
  const float* W2    = (const float*)d_in[6];
  const float* b2    = (const float*)d_in[7];
  const float* lng   = (const float*)d_in[8];
  const float* lnb   = (const float*)d_in[9];
  float* out = (float*)d_out;

  const int Nb = 64, S = 512, D = 1024, CH = 2048;

  char* ws = (char*)d_ws;
  size_t off = 0;
  auto alloc = [&](size_t bytes) -> char* {
    char* p = ws + off;
    off += (bytes + 255) & ~(size_t)255;
    return p;
  };
  __bf16* W1t  = (__bf16*)alloc((size_t)CH * (2 * D) * 2);        // [CH][2D] K-major
  __bf16* W2t  = (__bf16*)alloc((size_t)(4 * D) * CH * 2);        // [4D][CH] K-major
  __bf16* Wit  = (__bf16*)alloc((size_t)D * D * 2);               // [D][D]   K-major
  __bf16* xA   = (__bf16*)alloc((size_t)Nb * S * D * 2);          // 64 MB
  __bf16* xB   = (__bf16*)alloc((size_t)Nb * (S / 2) * D * 2);    // 32 MB
  __bf16* hbuf = (__bf16*)alloc((size_t)Nb * (S / 2) * CH * 2);   // 64 MB
  float*  cont = (float*) alloc((size_t)Nb * (S / 2) * 4 * D * 4);// 256 MB (>= init tmp)
  float*  mA   = (float*) alloc((size_t)Nb * S * 4);
  float*  mB   = (float*) alloc((size_t)Nb * (S / 2) * 4);

  dim3 tb(32, 8);
  // Weight transposes (fp32 [K][N] -> bf16 [N][K])
  transpose_convert<<<dim3(D / 32, D / 32), tb, 0, stream>>>(Wi, Wit, D, D);
  transpose_convert<<<dim3(CH / 32, (2 * D) / 32), tb, 0, stream>>>(W1, W1t, 2 * D, CH);
  transpose_convert<<<dim3((4 * D) / 32, CH / 32), tb, 0, stream>>>(W2, W2t, CH, 4 * D);

  // x0 = bf16(seq * mask); copy mask
  prep_x0<<<Nb * S, 256, 0, stream>>>(seq, imask, xA, D);
  copy_mask<<<(Nb * S + 255) / 256, 256, 0, stream>>>(imask, mA, Nb * S);

  // Init projection: tmp = x0 @ W_init + b_init  (fp32 out into `cont`)
  gemm_bf16_wmma<1><<<dim3(D / 128, (Nb * S) / 128), 256, 0, stream>>>(
      xA, Wit, bi, cont, Nb * S, D, D);
  // x = bf16(LN(tmp)*mask) back into xA
  ln_init<<<Nb * S, 256, 0, stream>>>(cont, mA, lng, lnb, xA, D);

  // Tree levels
  __bf16* xc = xA; __bf16* xn = xB;
  float*  mc = mA; float*  mn = mB;
  int Scur = S;
  while (Scur > 1) {
    int half = Scur / 2;
    int Mr   = Nb * half;
    // h = relu([l|r] @ W1 + b1): A is the xc buffer viewed as (Mr x 2D)
    gemm_bf16_wmma<0><<<dim3(CH / 128, (Mr + 127) / 128), 256, 0, stream>>>(
        xc, W1t, b1, hbuf, Mr, 2 * D, CH);
    // contents = h @ W2 + b2 (fp32)
    gemm_bf16_wmma<1><<<dim3((4 * D) / 128, (Mr + 127) / 128), 256, 0, stream>>>(
        hbuf, W2t, b2, cont, Mr, CH, 4 * D);
    float* outp = (half == 1) ? out : nullptr;
    compose<<<Mr, 256, 0, stream>>>(cont, xc, mc, lng, lnb, xn, mn, outp,
                                    D, Nb * D);
    { __bf16* t = xc; xc = xn; xn = t; }
    { float*  t = mc; mc = mn; mn = t; }
    Scur = half;
  }
}
